// MultinomialDDPM_58780922413561
// MI455X (gfx1250) — matrix operand order repaired and checked by reading, not compiled, not run on hardware
//
#include <hip/hip_runtime.h>
#include <hip/hip_bf16.h>
#include <math.h>

// ---------------- static problem config ----------------
#define BATCH   32768
#define FEAT    40
#define DCOLS   496
#define DPAD    512
#define TSTEPS  1000
#define L30     (-69.07755279f)        // logf(1e-30)
#define LN2     0.69314718055994531f
#define MBLOCKS (BATCH / 16)           // 2048 row tiles
#define KT_N    (DPAD / 32)            // 16 k-tiles of 32
#define NT_N    (DPAD / 16)            // 32 n-tiles of 16

typedef _Float16 h16;
typedef __attribute__((ext_vector_type(16))) _Float16 v16h;
typedef __attribute__((ext_vector_type(8)))  float    v8f;
typedef __attribute__((ext_vector_type(4)))  unsigned int v4u;
typedef __attribute__((ext_vector_type(8)))  int v8i_;
typedef __attribute__((ext_vector_type(4)))  int v4i_;

__device__ __forceinline__ float logaddexpf_(float a, float b) {
    float m = fmaxf(a, b);
    return m + log1pf(expf(fminf(a, b) - m));
}

// col -> feature. Features repeat sizes {2,4,8,16,32} (sum 62) eight times.
__device__ __forceinline__ int feat_of_col(int k) {
    int g = k / 62, lo = k % 62;
    int fl = (lo < 2) ? 0 : (lo < 6) ? 1 : (lo < 14) ? 2 : (lo < 30) ? 3 : 4;
    return 5 * g + fl;
}

// ---------------- kernel 0: zero scalar output ----------------
__global__ void k_zero(float* out) {
    if (blockIdx.x == 0 && threadIdx.x == 0) out[0] = 0.0f;
}

// ---------------- kernel 1: diffusion schedule tables ----------------
__global__ void k_init(float* la, float* l1a, float* lca, float* l1ca) {
    if (blockIdx.x == 0 && threadIdx.x == 0) {
        float acc = 0.0f;
        for (int t = 0; t < TSTEPS; ++t) {
            float beta = 1e-4f + (0.02f - 1e-4f) * ((float)t / (float)(TSTEPS - 1));
            float lg = logf(1.0f - beta);
            acc += lg;
            la[t]   = lg;
            l1a[t]  = logf(beta);
            lca[t]  = acc;
            l1ca[t] = logf(1.0f - expf(acc));
        }
    }
}

// ---------------- kernel 2: gumbel-max categorical sample per (row,feature) ----------------
__global__ void k_sample(const int* __restrict__ x0, const int* __restrict__ ts,
                         const float* __restrict__ uni,
                         const float* __restrict__ lca, const float* __restrict__ l1ca,
                         int* __restrict__ win) {
    int gid = blockIdx.x * blockDim.x + threadIdx.x;
    if (gid >= BATCH * FEAT) return;
    int b = gid / FEAT, f = gid % FEAT;
    int g = f / 5, fl = f % 5;
    int ncls = 2 << fl;                       // 2,4,8,16,32
    int s0 = 62 * g + (2 << fl) - 2;          // segment start column
    int t = ts[b];
    float ca = lca[t], m1 = l1ca[t];
    float lognc = (float)(fl + 1) * LN2;      // log(ncls)
    int w0 = x0[gid];                         // local class index of x0
    float zbest = -INFINITY;
    int cbest = 0;
    for (int c = 0; c < ncls; ++c) {
        float lx0   = (c == w0) ? 0.0f : L30;
        float logit = logaddexpf_(lx0 + ca, m1 - lognc);   // q_prior
        float u     = uni[(size_t)b * DCOLS + s0 + c];
        float gum   = -logf(-logf(u + 1e-30f) + 1e-30f);
        float z     = gum + logit;
        if (z > zbest) { zbest = z; cbest = c; }           // first-max tie break
    }
    win[gid] = s0 + cbest;                                 // winning global column
}

// ---------------- kernel 3: pack log_xt (f16) into WMMA A-fragment order ----------------
// A layout (ISA 7.12.2, 16-bit A 16x32): lane = h*16+m; VGPR v holds
// K = (v<4 ? 2v : 16+2(v-4)) + 8h and K+1.  -> k_local = (e<8 ? e : e+8) + 8h.
__global__ void k_packA(const int* __restrict__ win, h16* __restrict__ Ap) {
    int gid = blockIdx.x * blockDim.x + threadIdx.x;   // MBLOCKS*16*32 frags
    int mblk = gid >> 9;
    int rem  = gid & 511;
    int kt   = rem >> 5;
    int lane = rem & 31;
    int m = lane & 15, h = lane >> 4;
    int row = mblk * 16 + m;
    v16h va;
#pragma unroll
    for (int e = 0; e < 16; ++e) {
        int k = kt * 32 + ((e < 8) ? e : e + 8) + 8 * h;
        float v = 0.0f;
        if (k < DCOLS) {
            int f = feat_of_col(k);
            v = (win[(size_t)row * FEAT + f] == k) ? 0.0f : L30;
        }
        va[e] = (h16)v;
    }
    *(v16h*)(Ap + (size_t)gid * 16) = va;
}

// ---------------- kernel 4: pack W (f16) into WMMA B-fragment order ----------------
// B layout (32x16): lane = h*16+n; element e holds K = (e<8 ? e : e+8) + 8h of column n.
__global__ void k_packW(const float* __restrict__ W, h16* __restrict__ Wp) {
    int gid = blockIdx.x * blockDim.x + threadIdx.x;   // NT_N*KT_N*32 frags
    int nt   = gid >> 9;
    int rem  = gid & 511;
    int kt   = rem >> 5;
    int lane = rem & 31;
    int n = nt * 16 + (lane & 15);
    int h = lane >> 4;
    v16h vb;
#pragma unroll
    for (int e = 0; e < 16; ++e) {
        int k = kt * 32 + ((e < 8) ? e : e + 8) + 8 * h;
        float v = (k < DCOLS && n < DCOLS) ? W[(size_t)k * DCOLS + n] : 0.0f;
        vb[e] = (h16)v;
    }
    *(v16h*)(Wp + (size_t)gid * 16) = vb;
}

// ---------------- kernel 5: GEMM pred = log_xt @ W via v_wmma_f32_16x16x32_f16 ----------------
// Block = 256 threads = 8 waves. Block owns one 16-row tile, each wave 4 n-tiles (64 cols).
// A panel (16x512 f16 = 16KB, fragment order) staged in LDS via the Tensor Data Mover
// (tensor_load_to_lds + s_wait_tensorcnt), else cooperative b128 copy.
__global__ void __launch_bounds__(256) k_gemm(const h16* __restrict__ Ap,
                                              const h16* __restrict__ Wp,
                                              float* __restrict__ pred) {
    __shared__ __align__(16) h16 As[16 * DPAD];        // 16KB A panel, fragment order
    int tid  = threadIdx.x;
    int lane = tid & 31, wv = tid >> 5;
    int mblk = blockIdx.x;
    const h16* apanel = Ap + (size_t)mblk * (16 * DPAD);

#if __has_builtin(__builtin_amdgcn_tensor_load_to_lds) && __has_builtin(__builtin_amdgcn_s_wait_tensorcnt)
    // ---- TDM path: one descriptor moves the whole 16KB panel (1-D tile, 8192 x 2B) ----
    if (tid == 0) {
        unsigned long long ga = (unsigned long long)(const void*)apanel;
        unsigned lds = (unsigned)(uintptr_t)&As[0];
        // D# group0: count=1 | lds_addr | global_addr[56:0] | type=2 (bits 127:126)
        v4u g0 = { 1u,
                   lds,
                   (unsigned)(ga & 0xFFFFFFFFu),
                   (unsigned)((ga >> 32) & 0x01FFFFFFu) | 0x80000000u };
        // D# group1: mask=0, data_size=1(2B); tensor_dim0=8192 (bits79:48);
        // tensor_dim1=1 (bits111:80); tile_dim0=8192 (bits127:112);
        // tile_dim1/2=0; tensor_dim0_stride=8192 (bits207:160)
        v8i_ g1 = { 0x00010000,
                    0x20000000,      // tensor_dim0(8192) low16 << 16
                    0x00010000,      // tensor_dim0 hi=0 | tensor_dim1 lo=1
                    0x20000000,      // tensor_dim1 hi=0 | tile_dim0=8192
                    0,               // tile_dim1=0, tile_dim2=0
                    8192,            // tensor_dim0_stride low32
                    0, 0 };
        v4i_ g2 = { 0, 0, 0, 0 };
        v4i_ g3 = { 0, 0, 0, 0 };
        v8i_ g4 = { 0, 0, 0, 0, 0, 0, 0, 0 };   // extra descriptor slot (6-arg form)
        __builtin_amdgcn_tensor_load_to_lds(g0, g1, g2, g3, g4, 0);
        __builtin_amdgcn_s_wait_tensorcnt(0);
    }
    __syncthreads();
#else
    // ---- fallback: cooperative stage, 8192 halfs = 1024 uint4 ----
    {
        const uint4* src = (const uint4*)apanel;
        uint4* dst = (uint4*)As;
#pragma unroll
        for (int j = 0; j < 4; ++j) dst[tid * 4 + j] = src[tid * 4 + j];
    }
    __syncthreads();
#endif

    v8f acc[4];
#pragma unroll
    for (int j = 0; j < 4; ++j) acc[j] = (v8f){0.f, 0.f, 0.f, 0.f, 0.f, 0.f, 0.f, 0.f};

    // software pipeline: fragments for kt+1 are fetched before the WMMAs of kt
    v16h acur = *(const v16h*)&As[(0 * 32 + lane) * 16];
    v16h bcur[4];
#pragma unroll
    for (int j = 0; j < 4; ++j)
        bcur[j] = *(const v16h*)&Wp[((((size_t)wv * 4 + j) * 16 + 0) * 32 + lane) * 16];

#pragma unroll 1
    for (int kt = 0; kt < KT_N; ++kt) {
        v16h anxt = acur;
        v16h bnxt[4];
#pragma unroll
        for (int j = 0; j < 4; ++j) bnxt[j] = bcur[j];
        if (kt + 1 < KT_N) {
            anxt = *(const v16h*)&As[((kt + 1) * 32 + lane) * 16];
#pragma unroll
            for (int j = 0; j < 4; ++j)
                bnxt[j] = *(const v16h*)&Wp[((((size_t)wv * 4 + j) * 16 + (kt + 1)) * 32 + lane) * 16];
            __builtin_prefetch(&Wp[((((size_t)wv * 4) * 16 + kt + 1) * 32 + lane) * 16], 0, 1);
        }
#pragma unroll
        for (int j = 0; j < 4; ++j)
            acc[j] = __builtin_amdgcn_wmma_f32_16x16x32_f16(
                false, acur, false, bcur[j], (short)0, acc[j], false, false);
        acur = anxt;
#pragma unroll
        for (int j = 0; j < 4; ++j) bcur[j] = bnxt[j];
    }

    // C layout: VGPR r -> row r + 8h, col = lane&15 within the n-tile
    int h = lane >> 4, nl = lane & 15;
#pragma unroll
    for (int j = 0; j < 4; ++j) {
        int n = (wv * 4 + j) * 16 + nl;
        if (n < DCOLS) {
#pragma unroll
            for (int r = 0; r < 8; ++r) {
                int row = mblk * 16 + r + 8 * h;
                pred[(size_t)row * DCOLS + n] = acc[j][r];
            }
        }
    }
}

// ---------------- kernel 6: full loss epilogue, thread per (row,feature) ----------------
__global__ void __launch_bounds__(256) k_loss(
        const int* __restrict__ x0, const int* __restrict__ ts,
        const float* __restrict__ pred, const float* __restrict__ bias,
        const float* __restrict__ temb, const int* __restrict__ win,
        const float* __restrict__ la, const float* __restrict__ l1a,
        const float* __restrict__ lca, const float* __restrict__ l1ca,
        float* __restrict__ out) {
    __shared__ float red[256];
    int gid = blockIdx.x * blockDim.x + threadIdx.x;
    float contrib = 0.0f;
    if (gid < BATCH * FEAT) {
        int b = gid / FEAT, f = gid % FEAT;
        int g = f / 5, fl = f % 5;
        int ncls = 2 << fl;
        int s0 = 62 * g + (2 << fl) - 2;
        float lognc = (float)(fl + 1) * LN2;
        int t = ts[b];
        int tm1 = (t > 0) ? t - 1 : 0;
        float ca1 = lca[tm1], m1 = l1ca[tm1];
        float al  = la[t],   m1a = l1a[t];
        float caT = lca[TSTEPS - 1], mT = l1ca[TSTEPS - 1];
        int w0 = s0 + x0[gid];
        int wx = win[gid];

        float pv[32];
        // pass 1: denoiser logits + segment logsumexp (online)
        float mx = -INFINITY, sm = 0.0f;
        for (int c = 0; c < ncls; ++c) {
            int col = s0 + c;
            float v = pred[(size_t)b * DCOLS + col] + bias[col]
                    + temb[(size_t)t * DCOLS + col];
            pv[c] = v;
            if (v > mx) { sm = sm * expf(mx - v) + 1.0f; mx = v; }
            else        { sm += expf(v - mx); }
        }
        float lse_p = mx + logf(sm);

        // pass 2: un_true / un_est segment logsumexps
        float mtt = -INFINITY, stt = 0.0f, mee = -INFINITY, see = 0.0f;
        for (int c = 0; c < ncls; ++c) {
            int col = s0 + c;
            float lx0 = (col == w0) ? 0.0f : L30;
            float lxt = (col == wx) ? 0.0f : L30;
            float lxh = pv[c] - lse_p;                          // log_x0_hat
            float evt = (t == 0) ? lx0 : logaddexpf_(lx0 + ca1, m1 - lognc);
            float eve = (t == 0) ? lxh : logaddexpf_(lxh + ca1, m1 - lognc);
            float lt  = logaddexpf_(lxt + al, m1a - lognc);     // q_one_timestep
            float ut = evt + lt, ue = eve + lt;
            if (ut > mtt) { stt = stt * expf(mtt - ut) + 1.0f; mtt = ut; }
            else          { stt += expf(ut - mtt); }
            if (ue > mee) { see = see * expf(mee - ue) + 1.0f; mee = ue; }
            else          { see += expf(ue - mee); }
        }
        float lse_t = mtt + logf(stt), lse_e = mee + logf(see);

        // pass 3: KL / decoder NLL / prior KL
        float kl = 0.0f, dec = 0.0f, klp = 0.0f;
        for (int c = 0; c < ncls; ++c) {
            int col = s0 + c;
            float lx0 = (col == w0) ? 0.0f : L30;
            float lxt = (col == wx) ? 0.0f : L30;
            float lxh = pv[c] - lse_p;
            float evt = (t == 0) ? lx0 : logaddexpf_(lx0 + ca1, m1 - lognc);
            float eve = (t == 0) ? lxh : logaddexpf_(lxh + ca1, m1 - lognc);
            float lt  = logaddexpf_(lxt + al, m1a - lognc);
            float logT = (evt + lt) - lse_t;
            float logE = (eve + lt) - lse_e;
            kl  += expf(logT) * (logT - logE);
            dec -= expf(lx0) * logE;
            float qT = logaddexpf_(lx0 + caT, mT - lognc);
            klp += expf(qT) * (qT + lognc);
        }
        contrib = (((t == 0) ? dec : kl) * (float)TSTEPS + klp) * (1.0f / (float)BATCH);
    }
    red[threadIdx.x] = contrib;
    __syncthreads();
    for (int s = 128; s > 0; s >>= 1) {
        if (threadIdx.x < (unsigned)s) red[threadIdx.x] += red[threadIdx.x + s];
        __syncthreads();
    }
    if (threadIdx.x == 0) atomicAdd(out, red[0]);
}

// ---------------- host-side launch ----------------
extern "C" void kernel_launch(void* const* d_in, const int* in_sizes, int n_in,
                              void* d_out, int out_size, void* d_ws, size_t ws_size,
                              hipStream_t stream) {
    const int*   x0   = (const int*)d_in[0];   // [B,40]
    const int*   ts   = (const int*)d_in[1];   // [B]
    const float* uni  = (const float*)d_in[2]; // [B,496]
    const float* W    = (const float*)d_in[3]; // [496,496]
    const float* bias = (const float*)d_in[4]; // [496]
    const float* temb = (const float*)d_in[5]; // [1000,496]
    float* out = (float*)d_out;

    char* base = (char*)d_ws;
    float* la   = (float*)(base);                       // 4 x 1000 f32 tables
    float* l1a  = la + TSTEPS;
    float* lca  = l1a + TSTEPS;
    float* l1ca = lca + TSTEPS;
    size_t off = 16384;
    int* win  = (int*)(base + off);                     // [B,40] winners
    off += (size_t)BATCH * FEAT * sizeof(int);          // +5.24MB -> 5259264
    h16* Ap = (h16*)(base + off);                       // packed A frags, 33.5MB
    off += (size_t)MBLOCKS * 16 * DPAD * sizeof(h16);
    h16* Wp = (h16*)(base + off);                       // packed W frags, 0.5MB
    off += (size_t)DPAD * DPAD * sizeof(h16);
    float* pred = (float*)(base + off);                 // [B,496] f32, 65MB

    k_zero<<<1, 64, 0, stream>>>(out);
    k_init<<<1, 64, 0, stream>>>(la, l1a, lca, l1ca);
    k_sample<<<(BATCH * FEAT + 255) / 256, 256, 0, stream>>>(x0, ts, uni, lca, l1ca, win);
    k_packA<<<(MBLOCKS * 512) / 256, 256, 0, stream>>>(win, Ap);
    k_packW<<<(NT_N * 512) / 256, 256, 0, stream>>>(W, Wp);
    k_gemm<<<MBLOCKS, 256, 0, stream>>>(Ap, Wp, pred);
    k_loss<<<(BATCH * FEAT + 255) / 256, 256, 0, stream>>>(
        x0, ts, pred, bias, temb, win, la, l1a, lca, l1ca, out);
}
